// Causal_self_attention_78700980732404
// MI455X (gfx1250) — compile-verified, matrix operand order checked by reading, not run on hardware
//
#include <hip/hip_runtime.h>
#include <hip/hip_bf16.h>
#include <math.h>

// ---------------------------------------------------------------------------
// Causal self-attention, B=4 T=2048 C=768 H=12 D=64, fp32 in/out.
// cvt(f32->f16) -> QKV GEMM (async-LDS + WMMA) -> flash attention (WMMA)
//               -> proj GEMM (async-LDS + WMMA).
// GEMMs: 128x128 block tile (wave tile 32x64), double-buffered
// GLOBAL_LOAD_ASYNC_TO_LDS_B128 staging (ASYNCcnt-pipelined), ds_load_b128
// fragment loads, v_wmma_f32_16x16x32_f16 compute (8 WMMA / K-step / wave).
// ---------------------------------------------------------------------------

typedef __attribute__((ext_vector_type(16))) _Float16 v16h;
typedef __attribute__((ext_vector_type(8)))  _Float16 v8h;
typedef __attribute__((ext_vector_type(4)))  _Float16 v4h;
typedef __attribute__((ext_vector_type(8)))  float    v8f;

#define BQ 4
#define TQ 2048
#define CQ 768
#define HQ 12
#define DQ 64

__device__ __forceinline__ v8f wmma_f16(v16h a, v16h b, v8f c) {
  // (neg_a, A, neg_b, B, c_mod, C, reuse_a, reuse_b)
  return __builtin_amdgcn_wmma_f32_16x16x32_f16(false, a, false, b,
                                                (short)0, c, false, false);
}

// Async DMA: copy 16 bytes/lane global -> LDS, tracked by ASYNCcnt.
// GV mode: vdst = per-lane LDS byte address, vaddr = 64-bit global address.
__device__ __forceinline__ void async_b128(unsigned lds_addr,
                                           const _Float16* g) {
  asm volatile("global_load_async_to_lds_b128 %0, %1, off"
               :: "v"(lds_addr), "v"((unsigned long long)(uintptr_t)g)
               : "memory");
}

template <int N>
__device__ __forceinline__ void wait_asynccnt() {
  asm volatile("s_wait_asynccnt %0" :: "i"(N) : "memory");
}

// Load a 16x32 f16 A-fragment (or 32x16 B-fragment with lane==N) from a
// row-major tile with leading dimension `ld` halves.  CDNA5 16-bit A layout:
// lanes 0-15 = rows, K 0..7 (V0-3) + 16..23 (V4-7); lanes 16-31 = K 8..15,
// 24..31.
__device__ __forceinline__ v16h load_frag16(const _Float16* base, int ld) {
  const int lane = threadIdx.x & 31;
  const int r  = lane & 15;
  const int k0 = (lane >> 4) << 3;  // 0 or 8
  union { v16h v; v8h h[2]; } u;
  u.h[0] = *(const v8h*)(base + (size_t)r * ld + k0);
  u.h[1] = *(const v8h*)(base + (size_t)r * ld + k0 + 16);
  return u.v;
}

// ---------------------------------------------------------------------------
// f32 -> f16 convert, 4 elements / thread
// ---------------------------------------------------------------------------
__global__ void cvt_f32_f16(const float* __restrict__ in,
                            _Float16* __restrict__ out, int n4) {
  int i = blockIdx.x * blockDim.x + threadIdx.x;
  if (i < n4) {
    const float4 f = ((const float4*)in)[i];
    v4h h;
    h[0] = (_Float16)f.x; h[1] = (_Float16)f.y;
    h[2] = (_Float16)f.z; h[3] = (_Float16)f.w;
    ((v4h*)out)[i] = h;
  }
}

// ---------------------------------------------------------------------------
// Shared GEMM mainloop: C[128x128] += A[128xK] . B[128xK]^T for one block.
// 8 waves in a 4(M) x 2(N) grid; wave tile 32x64 (2x4 WMMA accumulators).
// A/B tiles staged through LDS with double-buffered async copies:
// 4 async_b128 per thread per K-step; s_wait_asynccnt 4 retires the older
// stage while the newer stage is in flight.
// ---------------------------------------------------------------------------
__device__ __forceinline__ void gemm_tile_128x128(
    const _Float16* __restrict__ A, const _Float16* __restrict__ B,
    int mBase, int nBase, int K, v8f acc[2][4]) {
  __shared__ _Float16 sA[2][128][32];   // 16 KB
  __shared__ _Float16 sB[2][128][32];   // 16 KB
  const int tid = threadIdx.x;
  const int wave = tid >> 5;
  const int wm = (wave & 3) * 32;
  const int wn = (wave >> 2) * 64;

  // Copy-chunk mapping: 16B chunks; A and B each 512 chunks -> 2/thread.
  const int r0 = tid >> 2;              // rows 0..63
  const int r1 = r0 + 64;               // rows 64..127
  const int kOff = (tid & 3) * 8;       // halves within the 32-wide K slab

  int cur = 0;
  {
    async_b128((unsigned)(uintptr_t)&sA[0][r0][kOff],
               A + (size_t)(mBase + r0) * K + kOff);
    async_b128((unsigned)(uintptr_t)&sA[0][r1][kOff],
               A + (size_t)(mBase + r1) * K + kOff);
    async_b128((unsigned)(uintptr_t)&sB[0][r0][kOff],
               B + (size_t)(nBase + r0) * K + kOff);
    async_b128((unsigned)(uintptr_t)&sB[0][r1][kOff],
               B + (size_t)(nBase + r1) * K + kOff);
  }
  for (int k = 0; k < K; k += 32) {
    if (k + 32 < K) {
      const int nxt = cur ^ 1;
      async_b128((unsigned)(uintptr_t)&sA[nxt][r0][kOff],
                 A + (size_t)(mBase + r0) * K + k + 32 + kOff);
      async_b128((unsigned)(uintptr_t)&sA[nxt][r1][kOff],
                 A + (size_t)(mBase + r1) * K + k + 32 + kOff);
      async_b128((unsigned)(uintptr_t)&sB[nxt][r0][kOff],
                 B + (size_t)(nBase + r0) * K + k + 32 + kOff);
      async_b128((unsigned)(uintptr_t)&sB[nxt][r1][kOff],
                 B + (size_t)(nBase + r1) * K + k + 32 + kOff);
      wait_asynccnt<4>();   // current stage's 4 copies are done
    } else {
      wait_asynccnt<0>();
    }
    __syncthreads();
    const v16h a0 = load_frag16(&sA[cur][wm][0], 32);
    const v16h a1 = load_frag16(&sA[cur][wm + 16][0], 32);
    v16h bf[4];
#pragma unroll
    for (int j = 0; j < 4; ++j)
      bf[j] = load_frag16(&sB[cur][wn + 16 * j][0], 32);
#pragma unroll
    for (int j = 0; j < 4; ++j) {
      acc[0][j] = wmma_f16(a0, bf[j], acc[0][j]);
      acc[1][j] = wmma_f16(a1, bf[j], acc[1][j]);
    }
    __syncthreads();        // finished reading `cur` before it is refilled
    cur ^= 1;
  }
}

// ---------------------------------------------------------------------------
// QKV GEMM: qkv[m, o] = x[m, :] . w_attn[o, :] + b_attn[o]
//   M = 8192, N = 2304, K = 768.  Epilogue scatters into K/Q/V (reference
// splits qkv as k,q,v) and folds the effective logit scale 2/sqrt(D)=0.25
// into Q.  Q,K layout [B,H,T,D]; V layout [B,H,D,T] (for PV B-fragments).
// ---------------------------------------------------------------------------
__global__ __launch_bounds__(256) void qkv_gemm(
    const _Float16* __restrict__ xh, const _Float16* __restrict__ wah,
    const float* __restrict__ b_attn,
    _Float16* __restrict__ qh, _Float16* __restrict__ kh,
    _Float16* __restrict__ vh) {
  const int lane = threadIdx.x & 31;
  const int wave = threadIdx.x >> 5;
  const int m0 = blockIdx.y * 128 + (wave & 3) * 32;
  const int n0 = blockIdx.x * 128 + (wave >> 2) * 64;

  v8f acc[2][4] = {};
  gemm_tile_128x128(xh, wah, blockIdx.y * 128, blockIdx.x * 128, CQ, acc);

  const int col = lane & 15;
  const int rowoff = (lane >> 4) << 3;  // C-layout: row = e + 8*(lane>=16)
#pragma unroll
  for (int i = 0; i < 2; ++i)
#pragma unroll
    for (int j = 0; j < 4; ++j)
#pragma unroll
      for (int e = 0; e < 8; ++e) {
        const int gm = m0 + 16 * i + e + rowoff;
        const int gn = n0 + 16 * j + col;
        const float v = acc[i][j][e] + b_attn[gn];
        const int sel = gn / CQ;        // 0=k, 1=q, 2=v (reference order)
        const int rem = gn - sel * CQ;
        const int h = rem >> 6, d = rem & 63;
        const int b = gm >> 11, t = gm & (TQ - 1);
        const size_t bh = (size_t)b * HQ + h;
        if (sel == 0)
          kh[(bh * TQ + t) * DQ + d] = (_Float16)v;
        else if (sel == 1)
          qh[(bh * TQ + t) * DQ + d] = (_Float16)(v * 0.25f);
        else
          vh[(bh * DQ + d) * TQ + t] = (_Float16)v;
      }
}

// ---------------------------------------------------------------------------
// Flash attention: block = (128 query rows, b*H+h); 8 waves x 16 rows.
// Q fragments resident; 32 keys/iter: 4 WMMA for S, online softmax via
// 16-lane shfl_xor row reductions, P re-fragmented through per-wave LDS,
// 4 WMMA for PV against transposed V.  Next K tile prefetched.
// ---------------------------------------------------------------------------
__global__ __launch_bounds__(256) void flash_attn(
    const _Float16* __restrict__ qh, const _Float16* __restrict__ kh,
    const _Float16* __restrict__ vh, _Float16* __restrict__ yh) {
  __shared__ _Float16 lds[8][16][40];  // per-wave 16x32 P tile, padded ld=40

  const int lane = threadIdx.x & 31;
  const int wave = threadIdx.x >> 5;
  const int bh = blockIdx.y;
  const int b = bh / HQ, h = bh - b * HQ;
  const int q0 = blockIdx.x * 128 + wave * 16;

  const _Float16* qp = qh + (size_t)bh * TQ * DQ;
  const _Float16* kp = kh + (size_t)bh * TQ * DQ;
  const _Float16* vp = vh + (size_t)bh * DQ * TQ;

  const v16h aQ0 = load_frag16(qp + (size_t)q0 * DQ, DQ);
  const v16h aQ1 = load_frag16(qp + (size_t)q0 * DQ + 32, DQ);

  const int col = lane & 15;
  const int rowoff = (lane >> 4) << 3;

  float rmax[8], rsum[8];
#pragma unroll
  for (int e = 0; e < 8; ++e) { rmax[e] = -__builtin_inff(); rsum[e] = 0.f; }
  v8f O[4] = {};

  for (int j = 0; j <= q0 + 15; j += 32) {
    // prefetch next key tile while computing this one
    __builtin_prefetch(kp + (size_t)(j + 32) * DQ, 0, 1);
    // ---- S = Q K^T (logits already scaled via Q) ----
    v8f S[2];
#pragma unroll
    for (int t2 = 0; t2 < 2; ++t2) {
      const v16h bk0 = load_frag16(kp + (size_t)(j + 16 * t2) * DQ, DQ);
      const v16h bk1 = load_frag16(kp + (size_t)(j + 16 * t2) * DQ + 32, DQ);
      v8f z = {};
      S[t2] = wmma_f16(aQ1, bk1, wmma_f16(aQ0, bk0, z));
    }
    // ---- causal mask ----
#pragma unroll
    for (int t2 = 0; t2 < 2; ++t2)
#pragma unroll
      for (int e = 0; e < 8; ++e) {
        const int kg = j + 16 * t2 + col;
        const int qg = q0 + e + rowoff;
        if (kg > qg) S[t2][e] = -__builtin_inff();
      }
    // ---- online softmax; row e+rowoff lives in one 16-lane half ----
#pragma unroll
    for (int e = 0; e < 8; ++e) {
      float mx = fmaxf(S[0][e], S[1][e]);
      mx = fmaxf(mx, __shfl_xor(mx, 1, 32));
      mx = fmaxf(mx, __shfl_xor(mx, 2, 32));
      mx = fmaxf(mx, __shfl_xor(mx, 4, 32));
      mx = fmaxf(mx, __shfl_xor(mx, 8, 32));
      const float mnew = fmaxf(rmax[e], mx);
      const float alpha = __expf(rmax[e] - mnew);
      rmax[e] = mnew;
      const float s0 = __expf(S[0][e] - mnew);
      const float s1 = __expf(S[1][e] - mnew);
      S[0][e] = s0; S[1][e] = s1;
      float rs = s0 + s1;
      rs += __shfl_xor(rs, 1, 32);
      rs += __shfl_xor(rs, 2, 32);
      rs += __shfl_xor(rs, 4, 32);
      rs += __shfl_xor(rs, 8, 32);
      rsum[e] = rsum[e] * alpha + rs;
#pragma unroll
      for (int c = 0; c < 4; ++c) O[c][e] *= alpha;
    }
    // ---- P: C-layout -> LDS -> A-fragment ----
#pragma unroll
    for (int t2 = 0; t2 < 2; ++t2)
#pragma unroll
      for (int e = 0; e < 8; ++e)
        lds[wave][e + rowoff][16 * t2 + col] = (_Float16)S[t2][e];
    const v16h pa = load_frag16(&lds[wave][0][0], 40);
    // ---- O += P V ; V stored [d, t] so B-fragment K-dim is contiguous ----
#pragma unroll
    for (int c = 0; c < 4; ++c) {
      const v16h bv = load_frag16(vp + (size_t)(16 * c) * TQ + j, TQ);
      O[c] = wmma_f16(pa, bv, O[c]);
    }
  }

  // ---- normalize and write y[b, t, h*64 + d] (f16) ----
#pragma unroll
  for (int c = 0; c < 4; ++c)
#pragma unroll
    for (int e = 0; e < 8; ++e) {
      const int row = q0 + e + rowoff;
      const int dc = h * DQ + 16 * c + col;
      const float v = O[c][e] / rsum[e];
      yh[((size_t)b * TQ + row) * CQ + dc] = (_Float16)v;
    }
}

// ---------------------------------------------------------------------------
// Projection GEMM: out[m, o] = y[m, :] . w_proj[o, :] + b_proj[o] (f32 out)
// M = 8192, N = 768, K = 768.
// ---------------------------------------------------------------------------
__global__ __launch_bounds__(256) void proj_gemm(
    const _Float16* __restrict__ yh, const _Float16* __restrict__ wph,
    const float* __restrict__ b_proj, float* __restrict__ out) {
  const int lane = threadIdx.x & 31;
  const int wave = threadIdx.x >> 5;
  const int m0 = blockIdx.y * 128 + (wave & 3) * 32;
  const int n0 = blockIdx.x * 128 + (wave >> 2) * 64;

  v8f acc[2][4] = {};
  gemm_tile_128x128(yh, wph, blockIdx.y * 128, blockIdx.x * 128, CQ, acc);

  const int col = lane & 15;
  const int rowoff = (lane >> 4) << 3;
#pragma unroll
  for (int i = 0; i < 2; ++i)
#pragma unroll
    for (int j = 0; j < 4; ++j)
#pragma unroll
      for (int e = 0; e < 8; ++e) {
        const int gm = m0 + 16 * i + e + rowoff;
        const int gn = n0 + 16 * j + col;
        out[(size_t)gm * CQ + gn] = acc[i][j][e] + b_proj[gn];
      }
}

// ---------------------------------------------------------------------------
// Launch
// ---------------------------------------------------------------------------
extern "C" void kernel_launch(void* const* d_in, const int* in_sizes, int n_in,
                              void* d_out, int out_size, void* d_ws,
                              size_t ws_size, hipStream_t stream) {
  (void)in_sizes; (void)n_in; (void)out_size; (void)ws_size;
  const float* x      = (const float*)d_in[0];
  const float* w_attn = (const float*)d_in[1];
  const float* b_attn = (const float*)d_in[2];
  const float* w_proj = (const float*)d_in[3];
  const float* b_proj = (const float*)d_in[4];
  float* out = (float*)d_out;

  const size_t NX  = (size_t)BQ * TQ * CQ;       // 6,291,456
  const size_t NWA = (size_t)3 * CQ * CQ;        // 1,769,472
  const size_t NWP = (size_t)CQ * CQ;            // 589,824

  _Float16* xh  = (_Float16*)d_ws;
  _Float16* wah = xh  + NX;
  _Float16* wph = wah + NWA;
  _Float16* qh  = wph + NWP;
  _Float16* kh  = qh  + NX;
  _Float16* vh  = kh  + NX;
  _Float16* yh  = vh  + NX;   // ~64.5 MB total f16 workspace

  cvt_f32_f16<<<(int)(NX / 1024), 256, 0, stream>>>(x, xh, (int)(NX / 4));
  cvt_f32_f16<<<(int)(NWA / 1024), 256, 0, stream>>>(w_attn, wah, (int)(NWA / 4));
  cvt_f32_f16<<<(int)(NWP / 1024), 256, 0, stream>>>(w_proj, wph, (int)(NWP / 4));

  qkv_gemm<<<dim3(18, 64), 256, 0, stream>>>(xh, wah, b_attn, qh, kh, vh);
  flash_attn<<<dim3(TQ / 128, BQ * HQ), 256, 0, stream>>>(qh, kh, vh, yh);
  proj_gemm<<<dim3(6, 64), 256, 0, stream>>>(yh, wph, b_proj, out);
}
